// encoder_attention_81810537054655
// MI455X (gfx1250) — matrix operand order repaired and checked by reading, not compiled, not run on hardware
//
#include <hip/hip_runtime.h>
#include <hip/hip_bf16.h>
#include <stddef.h>

// Problem dims (fixed by the reference)
#define B_ 32
#define N_ 2048
#define D_ 1024

typedef __bf16 bf16_t;
typedef bf16_t v16bf __attribute__((ext_vector_type(16)));
typedef float  v8f   __attribute__((ext_vector_type(8)));
typedef unsigned int uint4v __attribute__((ext_vector_type(4)));

// Pack two f32 into two bf16 (round-to-nearest-even) with one v_perm_b32.
__device__ __forceinline__ unsigned pack_bf16(float x, float y) {
  unsigned ux = __float_as_uint(x);
  unsigned uy = __float_as_uint(y);
  ux += 0x7FFFu + ((ux >> 16) & 1u);
  uy += 0x7FFFu + ((uy >> 16) & 1u);
  // result = { uy[31:16], ux[31:16] }
  return __builtin_amdgcn_perm(uy, ux, 0x07060302u);
}

// Branch-free tanh: 1 - 2*rcp(exp(2x)+1).  exp->inf => 1, exp->0 => -1.
// Uses v_exp_f32 + v_rcp_f32 (TRANS pipe, co-executes with VALU/WMMA).
__device__ __forceinline__ float tanh_fast(float x) {
  float t = __expf(2.0f * x);
  return 1.0f - 2.0f * __builtin_amdgcn_rcpf(t + 1.0f);
}

// ---------------------------------------------------------------------------
// Kernel 1: dec_fea[b,e] = sum_d st_hat[b,d] * W_s[e,d] + b_s[e]
// grid (D/256, B), block 256
// ---------------------------------------------------------------------------
__global__ void __launch_bounds__(256)
dec_fea_kernel(const float* __restrict__ st_hat, const float* __restrict__ W_s,
               const float* __restrict__ b_s, float* __restrict__ dec_fea) {
  const int b = blockIdx.y;
  const int e = blockIdx.x * 256 + threadIdx.x;
  const float4* ws = (const float4*)(W_s + (size_t)e * D_);
  const float4* sh = (const float4*)(st_hat + (size_t)b * D_);
  float acc = b_s[e];
#pragma unroll 4
  for (int i = 0; i < D_ / 4; ++i) {
    float4 a = ws[i], s = sh[i];
    acc += a.x * s.x + a.y * s.y + a.z * s.z + a.w * s.w;
  }
  dec_fea[b * D_ + e] = acc;
}

// ---------------------------------------------------------------------------
// Kernel 2: fused  et-GEMM -> tanh -> scores -> exp -> at_un/new_sum/denom
//   One workgroup per (b, 128-row n-tile). 8 waves in a 4(M)x2(N) grid;
//   each wave owns a 32x64 region = 2x4 WMMA 16x16 tiles (bf16, f32 acc).
//   Double-buffered LDS: one barrier per k-step, staging overlaps WMMA.
// grid (N/128, B), block 256
// ---------------------------------------------------------------------------
__global__ void __launch_bounds__(256)
attn_scores_kernel(const float* __restrict__ h, const float* __restrict__ W_h,
                   const float* __restrict__ dec_fea, const float* __restrict__ v,
                   const float* __restrict__ mask, const float* __restrict__ sum_ts,
                   float* __restrict__ at_un, float* __restrict__ denom_part,
                   float* __restrict__ new_sum_out) {
  const int b   = blockIdx.y;
  const int n0  = blockIdx.x * 128;
  const int tid = threadIdx.x;
  const int lane = tid & 31;
  const int w    = tid >> 5;
  const int wm   = w & 3;    // wave position along M (n rows): 4 waves * 32 rows
  const int wn   = w >> 2;   // wave position along N (e cols): 2 waves * 64 cols
  const int lm   = lane & 15;
  const int lh   = lane >> 4;

  // Double-buffered tiles; rows padded to 40 ushorts (80B, 16B-aligned).
  __shared__ __align__(16) unsigned short As[2][128][40]; // h tile   [m][k] bf16
  __shared__ __align__(16) unsigned short Bs[2][128][40]; // W_h tile [e][k] bf16
  __shared__ float sScores[128];

  if (tid < 128) sScores[tid] = 0.0f;

  float scoreAcc[2][8] = {};   // per M-subtile, per C-VGPR-row; lanes 0-15 and
                               // 16-31 hold the two 8-row halves of each tile
  const float* hB = h + (size_t)b * N_ * D_;

  // Stage one (et,kt) step's 128x32 tiles of h and W_h into LDS buffer `buf`.
  auto stage = [&](int s, int buf) {
    const int e0 = (s >> 5) * 128;
    const int k0 = (s & 31) * 32;
#pragma unroll
    for (int it = 0; it < 4; ++it) {
      const int lin = tid + it * 256;      // 0..1023 = 128 rows * 8 float4s
      const int row = lin >> 3;
      const int c4  = (lin & 7) * 4;
      const float4 av = *(const float4*)(hB + (size_t)(n0 + row) * D_ + k0 + c4);
      unsigned p0 = pack_bf16(av.x, av.y);
      unsigned p1 = pack_bf16(av.z, av.w);
      *(unsigned long long*)&As[buf][row][c4] =
          ((unsigned long long)p1 << 32) | p0;
      const float4 bv = *(const float4*)(W_h + (size_t)(e0 + row) * D_ + k0 + c4);
      unsigned q0 = pack_bf16(bv.x, bv.y);
      unsigned q1 = pack_bf16(bv.z, bv.w);
      *(unsigned long long*)&Bs[buf][row][c4] =
          ((unsigned long long)q1 << 32) | q0;
      // global_prefetch_b8: pull the k-after-next h slice toward the WGP
      __builtin_prefetch(hB + (size_t)(n0 + row) * D_ + ((k0 + 64) & (D_ - 1)) + c4, 0, 1);
    }
  };

  stage(0, 0);

  v8f acc[2][4];
  // Flattened loop: 8 e-tiles x 32 k-steps = 256 steps, double buffered.
  for (int s = 0; s < 256; ++s) {
    const int kt = s & 31;
    if (kt == 0) {
#pragma unroll
      for (int i = 0; i < 2; ++i)
#pragma unroll
        for (int j = 0; j < 4; ++j)
          acc[i][j] = (v8f){0.f, 0.f, 0.f, 0.f, 0.f, 0.f, 0.f, 0.f};
    }
    __syncthreads();                 // buffer s&1 is now fully staged
    if (s + 1 < 256) stage(s + 1, (s + 1) & 1);
    const int cur = s & 1;

    // A fragment (16x32): lane<16 -> K {0..7,16..23}; lane>=16 -> {8..15,24..31}
    union Frag { v16bf v; uint4v q[2]; };
    Frag a[2], bb[4];
#pragma unroll
    for (int i = 0; i < 2; ++i) {
      const unsigned short* p = &As[cur][wm * 32 + i * 16 + lm][lh * 8];
      a[i].q[0] = *(const uint4v*)p;        // K = lh*8 .. +7
      a[i].q[1] = *(const uint4v*)(p + 16); // K = 16+lh*8 .. +7
    }
    // B fragment (32x16): lane n=lm holds K = lh*8 .. +7, 16+lh*8 .. +7
#pragma unroll
    for (int j = 0; j < 4; ++j) {
      const unsigned short* p = &Bs[cur][wn * 64 + j * 16 + lm][lh * 8];
      bb[j].q[0] = *(const uint4v*)p;
      bb[j].q[1] = *(const uint4v*)(p + 16);
    }
#pragma unroll
    for (int i = 0; i < 2; ++i)
#pragma unroll
      for (int j = 0; j < 4; ++j)
        acc[i][j] = __builtin_amdgcn_wmma_f32_16x16x32_bf16(
            false, a[i].v, false, bb[j].v, (short)0, acc[i][j], false, false);

    if (kt == 31) {
      // Epilogue for this e-tile: score += v[e] * tanh(et + dec_fea[b,e])
      const int e0 = (s >> 5) * 128;
#pragma unroll
      for (int j = 0; j < 4; ++j) {
        const int e = e0 + wn * 64 + j * 16 + lm;   // C-matrix col = lane%16
        const float df = dec_fea[b * D_ + e];
        const float vv = v[e];
#pragma unroll
        for (int i = 0; i < 2; ++i)
#pragma unroll
          for (int r = 0; r < 8; ++r)
            scoreAcc[i][r] += vv * tanh_fast(acc[i][j][r] + df);
      }
    }
  }

  // Reduce across the 16 lanes of each C half; combine 2 e-waves via LDS atomics.
#pragma unroll
  for (int i = 0; i < 2; ++i)
#pragma unroll
    for (int r = 0; r < 8; ++r) {
      float s = scoreAcc[i][r];
      s += __shfl_xor(s, 1);
      s += __shfl_xor(s, 2);
      s += __shfl_xor(s, 4);
      s += __shfl_xor(s, 8);
      if (lane == 0)  atomicAdd(&sScores[wm * 32 + i * 16 + r], s);
      if (lane == 16) atomicAdd(&sScores[wm * 32 + i * 16 + 8 + r], s);
    }
  __syncthreads();

  if (tid < 128) {
    const int n = n0 + tid;
    const float sc  = sScores[tid];
    const float ex  = expf(sc);
    const float sts = sum_ts[b * N_ + n];
    const float au  = (ex / sts) * mask[b * N_ + n]; // unnormalized attention
    at_un[b * N_ + n] = au;
    new_sum_out[b * N_ + n] = sts + ex;
    sScores[tid] = au;
  }
  __syncthreads();
  if (tid < 64) sScores[tid] += sScores[tid + 64];
  __syncthreads();
  if (tid < 32) {
    float x = sScores[tid] + sScores[tid + 32];
    x += __shfl_xor(x, 1);
    x += __shfl_xor(x, 2);
    x += __shfl_xor(x, 4);
    x += __shfl_xor(x, 8);
    x += __shfl_xor(x, 16);
    if (tid == 0) denom_part[b * gridDim.x + blockIdx.x] = x;
  }
}

// ---------------------------------------------------------------------------
// Kernel 3: denom[b] = sum of 16 tile partials.  grid(B), block 32
// ---------------------------------------------------------------------------
__global__ void __launch_bounds__(32)
denom_kernel(const float* __restrict__ denom_part, float* __restrict__ denom) {
  const int b = blockIdx.x;
  const int l = threadIdx.x;
  float x = (l < 16) ? denom_part[b * 16 + l] : 0.0f;
  x += __shfl_xor(x, 1);
  x += __shfl_xor(x, 2);
  x += __shfl_xor(x, 4);
  x += __shfl_xor(x, 8);
  if (l == 0) denom[b] = x;
}

// ---------------------------------------------------------------------------
// Kernel 4: at = at_un / denom ; ct_e[b,d] = sum_n at[b,n] * h[b,n,d]
// grid (D/256, B), block 256 (each block also normalizes a 512-wide at chunk)
// ---------------------------------------------------------------------------
__global__ void __launch_bounds__(256)
ctx_kernel(const float* __restrict__ h, const float* __restrict__ at_un,
           const float* __restrict__ denom, float* __restrict__ ct_out,
           float* __restrict__ at_out) {
  const int b = blockIdx.y;
  const int t = threadIdx.x;
  const int d = blockIdx.x * 256 + t;
  const float inv = 1.0f / denom[b];

  const int nbase = blockIdx.x * 512;  // gridDim.x = 4, covers N = 2048
  at_out[b * N_ + nbase + t]       = at_un[b * N_ + nbase + t] * inv;
  at_out[b * N_ + nbase + 256 + t] = at_un[b * N_ + nbase + 256 + t] * inv;

  const float* hb = h + (size_t)b * N_ * D_ + d;
  const float* au = at_un + (size_t)b * N_;
  float a0 = 0.f, a1 = 0.f, a2 = 0.f, a3 = 0.f;
  for (int n = 0; n < N_; n += 4) {
    a0 += au[n + 0] * hb[(size_t)(n + 0) * D_];
    a1 += au[n + 1] * hb[(size_t)(n + 1) * D_];
    a2 += au[n + 2] * hb[(size_t)(n + 2) * D_];
    a3 += au[n + 3] * hb[(size_t)(n + 3) * D_];
  }
  ct_out[b * D_ + d] = (a0 + a1 + a2 + a3) * inv;
}

// ---------------------------------------------------------------------------
extern "C" void kernel_launch(void* const* d_in, const int* in_sizes, int n_in,
                              void* d_out, int out_size, void* d_ws, size_t ws_size,
                              hipStream_t stream) {
  const float* st_hat = (const float*)d_in[0]; // [B,D]
  const float* h      = (const float*)d_in[1]; // [B,N,D]
  const float* mask   = (const float*)d_in[2]; // [B,N]
  const float* sum_ts = (const float*)d_in[3]; // [B,N]
  const float* W_h    = (const float*)d_in[4]; // [D,D]
  const float* W_s    = (const float*)d_in[5]; // [D,D]
  const float* b_s    = (const float*)d_in[6]; // [D]
  const float* v      = (const float*)d_in[7]; // [D]

  float* out      = (float*)d_out;
  float* ct_e     = out;                       // B*D
  float* at_out   = out + B_ * D_;             // B*N
  float* new_sum  = out + B_ * D_ + B_ * N_;   // B*N

  float* ws         = (float*)d_ws;
  float* dec_fea    = ws;                      // B*D
  float* at_un      = dec_fea + B_ * D_;       // B*N
  float* denom_part = at_un + B_ * N_;         // B*16
  float* denom      = denom_part + B_ * 16;    // B

  dec_fea_kernel<<<dim3(D_ / 256, B_), 256, 0, stream>>>(st_hat, W_s, b_s, dec_fea);
  attn_scores_kernel<<<dim3(N_ / 128, B_), 256, 0, stream>>>(
      h, W_h, dec_fea, v, mask, sum_ts, at_un, denom_part, new_sum);
  denom_kernel<<<dim3(B_), 32, 0, stream>>>(denom_part, denom);
  ctx_kernel<<<dim3(D_ / 256, B_), 256, 0, stream>>>(h, at_un, denom, ct_e, at_out);
}
